// TransformerBlock_24266565223074
// MI455X (gfx1250) — matrix operand order repaired and checked by reading, not compiled, not run on hardware
//
#include <hip/hip_runtime.h>
#include <hip/hip_bf16.h>
#include <math.h>

// ---------------------------------------------------------------------------
// Types / helpers
// ---------------------------------------------------------------------------
typedef __attribute__((ext_vector_type(16))) __bf16 v16bf;
typedef __attribute__((ext_vector_type(8)))  float  v8f;
typedef __attribute__((__vector_size__(16))) int    v4i;

#define WMMA_BF16(a, b, c) \
    __builtin_amdgcn_wmma_f32_16x16x32_bf16(false, (a), false, (b), (short)0, (c), false, false)

__device__ __forceinline__ unsigned short f2bf(float f) {
    union { float f; unsigned int u; } a; a.f = f;
    unsigned int u = a.u;
    unsigned int r = u + 0x7fffu + ((u >> 16) & 1u);   // round-to-nearest-even
    return (unsigned short)(r >> 16);
}

// ---------------------------------------------------------------------------
// Async global->LDS staging (CDNA5 GLOBAL_LOAD_ASYNC_TO_LDS_B128, ASYNCcnt),
// with a synchronous uint4 fallback if the builtin is unavailable.
// ROCm clang types the builtin as (v4i addrspace(1)*, v4i addrspace(3)*, imm, imm).
// ---------------------------------------------------------------------------
#if defined(__HIP_DEVICE_COMPILE__) && __has_builtin(__builtin_amdgcn_global_load_async_to_lds_b128)
#define ASYNC_COPY 1
#else
#define ASYNC_COPY 0
#endif

__device__ __forceinline__ void g2l_b128(const unsigned short* g, unsigned short* l) {
#if ASYNC_COPY
    typedef __attribute__((address_space(1))) v4i gv4i;
    typedef __attribute__((address_space(3))) v4i lv4i;
    __builtin_amdgcn_global_load_async_to_lds_b128(
        (gv4i*)(unsigned long long)(size_t)g,
        (lv4i*)(unsigned int)(size_t)l,
        0, 0);
#else
    *(uint4*)l = *(const uint4*)g;
#endif
}

__device__ __forceinline__ void async_fence() {
#if ASYNC_COPY
#if __has_builtin(__builtin_amdgcn_s_wait_asynccnt)
    __builtin_amdgcn_s_wait_asynccnt(0);
#else
    asm volatile("s_wait_asynccnt 0" ::: "memory");
#endif
#endif
}

// ---------------------------------------------------------------------------
// Fragment loaders. LDS pitches are multiples of 8 ushorts so every A-fragment
// is 2 x ds_load_b128.
// A-fragment (16x32 bf16, [m][k] tile): lane m=lane&15, hi=lane>>4 holds
// k = 8*hi..8*hi+7 (bytes [hi*16, hi*16+16)) and k = 16+8*hi..+7 (+32B).
// ---------------------------------------------------------------------------
__device__ __forceinline__ v16bf load_a_frag16(const unsigned short* base, int lane, int pitch) {
    int m = lane & 15, hi = lane >> 4;
    union { v16bf v; uint4 q[2]; } f;
    const uint4* p = (const uint4*)(base + m * pitch + hi * 8);
    f.q[0] = p[0];
    f.q[1] = p[2];
    return f.v;
}

// B-fragment (32x16 bf16, [k][n] tile): lane n=lane&15, hi=lane>>4 holds
// k = 8*hi..8*hi+7 and k = 16+8*hi..+7 (same order as the A-fragment).
__device__ __forceinline__ v16bf load_b_frag16(const unsigned short* base, int lane, int pitch, int n0) {
    int n = lane & 15, hi = lane >> 4;
    union { v16bf v; unsigned short u[16]; } f;
#pragma unroll
    for (int s = 0; s < 8; ++s) {
        f.u[s]     = base[(8 * hi + s) * pitch + n0 + n];
        f.u[8 + s] = base[(16 + 8 * hi + s) * pitch + n0 + n];
    }
    return f.v;
}

// ---------------------------------------------------------------------------
// fp32 [N,K] -> bf16 transposed [K,N] (done once per call per weight so the
// GEMM B-tiles are natural row copies; keeps epilogue stores coalesced).
// ---------------------------------------------------------------------------
__global__ __launch_bounds__(256) void cvt_t_bf16_kernel(const float* __restrict__ in,
                                                         unsigned short* __restrict__ out,
                                                         int N, int K) {
    size_t boff = (size_t)blockIdx.z * N * K;
    in  += boff; out += boff;
    __shared__ unsigned short t[32][33];
    int k0 = blockIdx.x * 32, n0 = blockIdx.y * 32;
    int tx = threadIdx.x & 31, ty = threadIdx.x >> 5;
    for (int i = ty; i < 32; i += 8)
        t[i][tx] = f2bf(in[(size_t)(n0 + i) * K + k0 + tx]);
    __syncthreads();
    for (int i = ty; i < 32; i += 8)
        out[(size_t)(k0 + i) * N + n0 + tx] = t[tx][i];
}

// ---------------------------------------------------------------------------
// LayerNorm (fp32 in, bf16 out), one row per block
// ---------------------------------------------------------------------------
__global__ __launch_bounds__(256) void ln_bf16_kernel(const float* __restrict__ x,
                                                      const float* __restrict__ g,
                                                      const float* __restrict__ be,
                                                      unsigned short* __restrict__ out, int D) {
    int row = blockIdx.x;
    const float* xr = x + (size_t)row * D;
    float s = 0.f, s2 = 0.f;
    for (int i = threadIdx.x; i < D; i += blockDim.x) { float v = xr[i]; s += v; s2 += v * v; }
#pragma unroll
    for (int m = 1; m < 32; m <<= 1) { s += __shfl_xor(s, m, 32); s2 += __shfl_xor(s2, m, 32); }
    __shared__ float reds[8][2];
    __shared__ float stat[2];
    int wid = threadIdx.x >> 5, lane = threadIdx.x & 31;
    if (lane == 0) { reds[wid][0] = s; reds[wid][1] = s2; }
    __syncthreads();
    if (threadIdx.x == 0) {
        float t = 0.f, t2 = 0.f;
#pragma unroll
        for (int i = 0; i < 8; ++i) { t += reds[i][0]; t2 += reds[i][1]; }
        float mu  = t / (float)D;
        float var = t2 / (float)D - mu * mu;
        stat[0] = mu; stat[1] = rsqrtf(var + 1e-5f);
    }
    __syncthreads();
    float mu = stat[0], inv = stat[1];
    for (int i = threadIdx.x; i < D; i += blockDim.x)
        out[(size_t)row * D + i] = f2bf((xr[i] - mu) * inv * g[i] + be[i]);
}

// ---------------------------------------------------------------------------
// Per-head QKV projection: per head h, y = hh @ W^T + b  (K = 64).
// Weights arrive pre-transposed: Wt[h] is [64(k)][64(n)] bf16.
// grid: (M/128, H), block 256 (8 waves); each wave owns one 16-row m-tile.
// Output layout: [B*H, S, 64] bf16; 1/sqrt(HD)=0.125 folded into Q.
// ---------------------------------------------------------------------------
__global__ __launch_bounds__(256) void qkv_kernel(
    const unsigned short* __restrict__ hbf,                        // [8192,1024]
    const unsigned short* __restrict__ Wqt,
    const unsigned short* __restrict__ Wkt,
    const unsigned short* __restrict__ Wvt,                        // [16][64][64] k-major
    const float* __restrict__ bq, const float* __restrict__ bk, const float* __restrict__ bv,
    unsigned short* __restrict__ q, unsigned short* __restrict__ k, unsigned short* __restrict__ v) {
    const int D = 1024, HD = 64, S = 2048, H = 16;
    int h  = blockIdx.y;
    int m0 = blockIdx.x * 128;
    __shared__ unsigned short As[128][72];
    __shared__ unsigned short Bq[64][72], Bk[64][72], Bv[64][72];
    int tid = threadIdx.x, lane = tid & 31, w = tid >> 5;

    for (int c = tid; c < 128 * 8; c += 256) {                     // A tile (async)
        int r = c >> 3, c8 = (c & 7) * 8;
        g2l_b128(hbf + (size_t)(m0 + r) * D + h * HD + c8, &As[r][c8]);
    }
    for (int c = tid; c < 64 * 8; c += 256) {                      // W tiles (async, no transpose)
        int kr = c >> 3, n8 = (c & 7) * 8;
        g2l_b128(Wqt + (size_t)h * 4096 + kr * 64 + n8, &Bq[kr][n8]);
        g2l_b128(Wkt + (size_t)h * 4096 + kr * 64 + n8, &Bk[kr][n8]);
        g2l_b128(Wvt + (size_t)h * 4096 + kr * 64 + n8, &Bv[kr][n8]);
    }
    async_fence();
    __syncthreads();

    int mt = w * 16;
    v8f accq[4] = {}, acck[4] = {}, accv[4] = {};
#pragma unroll
    for (int ks = 0; ks < 2; ++ks) {
        v16bf a = load_a_frag16(&As[mt][ks * 32], lane, 72);
#pragma unroll
        for (int ni = 0; ni < 4; ++ni) {
            accq[ni] = WMMA_BF16(a, load_b_frag16(&Bq[ks * 32][0], lane, 72, ni * 16), accq[ni]);
            acck[ni] = WMMA_BF16(a, load_b_frag16(&Bk[ks * 32][0], lane, 72, ni * 16), acck[ni]);
            accv[ni] = WMMA_BF16(a, load_b_frag16(&Bv[ks * 32][0], lane, 72, ni * 16), accv[ni]);
        }
    }
    int hi = lane >> 4, ln = lane & 15;
#pragma unroll
    for (int ni = 0; ni < 4; ++ni) {
        int n = ni * 16 + ln;
#pragma unroll
        for (int r = 0; r < 8; ++r) {
            int m = m0 + mt + r + 8 * hi;
            int b_ = m >> 11, s_ = m & 2047;
            size_t dst = (((size_t)(b_ * H + h)) * S + s_) * HD + n;
            q[dst] = f2bf((accq[ni][r] + bq[h * HD + n]) * 0.125f);
            k[dst] = f2bf( acck[ni][r] + bk[h * HD + n]);
            v[dst] = f2bf( accv[ni][r] + bv[h * HD + n]);
        }
    }
}

// ---------------------------------------------------------------------------
// Causal flash attention. grid: (S/64, B*H), block 128 (4 waves, 16 rows each).
// Output: attn bf16 in [B*S, D] layout (col = h*64+d) for the out-proj GEMM.
// ---------------------------------------------------------------------------
__global__ __launch_bounds__(128) void attn_kernel(
    const unsigned short* __restrict__ q, const unsigned short* __restrict__ k,
    const unsigned short* __restrict__ v, unsigned short* __restrict__ attn) {
    const int S = 2048, HD = 64, D = 1024;
    int bh = blockIdx.y, qb = blockIdx.x, qr0 = qb * 64;
    const unsigned short* qbase = q + (size_t)bh * S * HD;
    const unsigned short* kbase = k + (size_t)bh * S * HD;
    const unsigned short* vbase = v + (size_t)bh * S * HD;

    __shared__ unsigned short Qs[64][72], Ks[64][72], Vs[64][72];
    __shared__ unsigned short Ps[4][16][72];
    int tid = threadIdx.x, lane = tid & 31, w = tid >> 5;
    int hi = lane >> 4, ln = lane & 15;

    for (int c = tid; c < 64 * 8; c += 128) {
        int r = c >> 3, c8 = (c & 7) * 8;
        g2l_b128(qbase + (size_t)(qr0 + r) * HD + c8, &Qs[r][c8]);
    }
    async_fence();
    __syncthreads();
    v16bf aq0 = load_a_frag16(&Qs[w * 16][0],  lane, 72);
    v16bf aq1 = load_a_frag16(&Qs[w * 16][32], lane, 72);

    v8f Of[4] = {};
    float rmax[8], rsum[8];
#pragma unroll
    for (int r = 0; r < 8; ++r) { rmax[r] = -3.0e38f; rsum[r] = 0.f; }

    for (int kb = 0; kb <= qb; ++kb) {
        int kr0 = kb * 64;
        __syncthreads();
        for (int c = tid; c < 64 * 8; c += 128) {
            int r = c >> 3, c8 = (c & 7) * 8;
            union { uint4 q4; unsigned short us[8]; } u;       // K: transpose scatter
            u.q4 = *(const uint4*)(kbase + (size_t)(kr0 + r) * HD + c8);
#pragma unroll
            for (int t = 0; t < 8; ++t) Ks[c8 + t][r] = u.us[t];
            g2l_b128(vbase + (size_t)(kr0 + r) * HD + c8, &Vs[r][c8]);   // V: async
        }
        async_fence();
        __syncthreads();

        // S = Q K^T  (16x64 per wave)
        v8f sf[4];
#pragma unroll
        for (int ni = 0; ni < 4; ++ni) {
            v8f acc = {};
            acc = WMMA_BF16(aq0, load_b_frag16(&Ks[0][0],  lane, 72, ni * 16), acc);
            acc = WMMA_BF16(aq1, load_b_frag16(&Ks[32][0], lane, 72, ni * 16), acc);
            sf[ni] = acc;
        }
        if (kb == qb) {                         // diagonal block: causal mask
#pragma unroll
            for (int ni = 0; ni < 4; ++ni) {
                int n_g = kr0 + ni * 16 + ln;
#pragma unroll
                for (int r = 0; r < 8; ++r) {
                    int m_g = qr0 + w * 16 + r + 8 * hi;
                    if (n_g > m_g) sf[ni][r] = -3.0e38f;
                }
            }
        }
        // online softmax (row = r + 8*hi, replicated across 16 lanes)
#pragma unroll
        for (int r = 0; r < 8; ++r) {
            float mloc = fmaxf(fmaxf(sf[0][r], sf[1][r]), fmaxf(sf[2][r], sf[3][r]));
#pragma unroll
            for (int mk = 1; mk < 16; mk <<= 1) mloc = fmaxf(mloc, __shfl_xor(mloc, mk, 32));
            float mnew  = fmaxf(rmax[r], mloc);
            float scale = __expf(rmax[r] - mnew);
            float psum  = 0.f;
#pragma unroll
            for (int ni = 0; ni < 4; ++ni) {
                float p = __expf(sf[ni][r] - mnew);
                sf[ni][r] = p; psum += p;
            }
#pragma unroll
            for (int mk = 1; mk < 16; mk <<= 1) psum += __shfl_xor(psum, mk, 32);
            rsum[r] = rsum[r] * scale + psum;
            rmax[r] = mnew;
#pragma unroll
            for (int ni = 0; ni < 4; ++ni) Of[ni][r] *= scale;
        }
        // P (bf16) into per-wave LDS in A-tile order, then O += P * V
#pragma unroll
        for (int ni = 0; ni < 4; ++ni)
#pragma unroll
            for (int r = 0; r < 8; ++r)
                Ps[w][r + 8 * hi][ni * 16 + ln] = f2bf(sf[ni][r]);
        __syncthreads();
        v16bf ap0 = load_a_frag16(&Ps[w][0][0],  lane, 72);
        v16bf ap1 = load_a_frag16(&Ps[w][0][32], lane, 72);
#pragma unroll
        for (int ni = 0; ni < 4; ++ni) {
            Of[ni] = WMMA_BF16(ap0, load_b_frag16(&Vs[0][0],  lane, 72, ni * 16), Of[ni]);
            Of[ni] = WMMA_BF16(ap1, load_b_frag16(&Vs[32][0], lane, 72, ni * 16), Of[ni]);
        }
    }

    int b_ = bh >> 4, h_ = bh & 15;
#pragma unroll
    for (int r = 0; r < 8; ++r) {
        float inv = 1.0f / rsum[r];
        int m_g = b_ * S + qr0 + w * 16 + r + 8 * hi;
#pragma unroll
        for (int ni = 0; ni < 4; ++ni) {
            int d_ = ni * 16 + ln;
            attn[(size_t)m_g * D + h_ * HD + d_] = f2bf(Of[ni][r] * inv);
        }
    }
}

// ---------------------------------------------------------------------------
// Generic bf16 GEMM: out = epi(A @ W^T + bias) [+ res]
// A [M,K] bf16 row-major; Wt [K,N] bf16 (pre-transposed) -> B tiles are
// natural row copies. Block tile 128x256, K stages of 32, double-buffered
// async staging, 8 waves (2m x 4n), wave tile 64x64 = 16 WMMAs per K-step.
// ---------------------------------------------------------------------------
template <int GELU_EPI, bool OUT_BF, bool HAS_RES>
__global__ __launch_bounds__(256) void gemm_kernel(
    const unsigned short* __restrict__ A, const unsigned short* __restrict__ Wt,
    const float* __restrict__ bias, const float* __restrict__ res,
    void* __restrict__ outp, int M, int N, int K) {
    __shared__ unsigned short As[2][128][40];
    __shared__ unsigned short Bs[2][32][264];
    int tid = threadIdx.x, lane = tid & 31, w = tid >> 5;
    int wm = w >> 2, wn = w & 3;
    int m0 = blockIdx.x * 128, n0 = blockIdx.y * 256;
    v8f acc[4][4] = {};
    int nK = K >> 5;

    auto load_tile = [&](int buf, int k0) {
#pragma unroll
        for (int it = 0; it < 2; ++it) {                        // A: 128x32
            int c = tid + it * 256;
            int r = c >> 2, c8 = (c & 3) * 8;
            g2l_b128(A + (size_t)(m0 + r) * K + k0 + c8, &As[buf][r][c8]);
        }
#pragma unroll
        for (int it = 0; it < 4; ++it) {                        // B: 32x256 (natural)
            int c = tid + it * 256;
            int kr = c >> 5, n8 = (c & 31) * 8;
            g2l_b128(Wt + (size_t)(k0 + kr) * N + n0 + n8, &Bs[buf][kr][n8]);
        }
    };

    load_tile(0, 0);
    async_fence();
    __syncthreads();
    for (int kt = 0; kt < nK; ++kt) {
        int cur = kt & 1;
        if (kt + 1 < nK) load_tile(cur ^ 1, (kt + 1) << 5);
        v16bf af[4], bfv[4];
#pragma unroll
        for (int mi = 0; mi < 4; ++mi)
            af[mi] = load_a_frag16(&As[cur][wm * 64 + mi * 16][0], lane, 40);
#pragma unroll
        for (int ni = 0; ni < 4; ++ni)
            bfv[ni] = load_b_frag16(&Bs[cur][0][0], lane, 264, wn * 64 + ni * 16);
#pragma unroll
        for (int mi = 0; mi < 4; ++mi)
#pragma unroll
            for (int ni = 0; ni < 4; ++ni)
                acc[mi][ni] = WMMA_BF16(af[mi], bfv[ni], acc[mi][ni]);
        async_fence();
        __syncthreads();
    }

    int hi = lane >> 4, ln = lane & 15;
#pragma unroll
    for (int mi = 0; mi < 4; ++mi)
#pragma unroll
        for (int ni = 0; ni < 4; ++ni) {
            int col = n0 + wn * 64 + ni * 16 + ln;
#pragma unroll
            for (int r = 0; r < 8; ++r) {
                int row = m0 + wm * 64 + mi * 16 + r + 8 * hi;
                float val = acc[mi][ni][r] + bias[col];
                if (GELU_EPI) val = 0.5f * val * (1.0f + erff(val * 0.70710678118f));
                if (HAS_RES) val += res[(size_t)row * N + col];
                if (OUT_BF) ((unsigned short*)outp)[(size_t)row * N + col] = f2bf(val);
                else        ((float*)outp)[(size_t)row * N + col] = val;
            }
        }
}

// ---------------------------------------------------------------------------
// Host launcher
// ---------------------------------------------------------------------------
extern "C" void kernel_launch(void* const* d_in, const int* in_sizes, int n_in,
                              void* d_out, int out_size, void* d_ws, size_t ws_size,
                              hipStream_t stream) {
    const int B = 4, S = 2048, D = 1024, H = 16, HD = 64;
    const int M = B * S;                                   // 8192
    const float* x   = (const float*)d_in[0];
    const float* g1  = (const float*)d_in[1];
    const float* be1 = (const float*)d_in[2];
    const float* Wq  = (const float*)d_in[3];
    const float* bq  = (const float*)d_in[4];
    const float* Wk  = (const float*)d_in[5];
    const float* bk  = (const float*)d_in[6];
    const float* Wv  = (const float*)d_in[7];
    const float* bv  = (const float*)d_in[8];
    const float* Wp  = (const float*)d_in[9];
    const float* bp  = (const float*)d_in[10];
    const float* g2  = (const float*)d_in[11];
    const float* be2 = (const float*)d_in[12];
    const float* W1  = (const float*)d_in[13];
    const float* b1  = (const float*)d_in[14];
    const float* W2  = (const float*)d_in[15];
    const float* b2  = (const float*)d_in[16];
    float* out = (float*)d_out;

    char* ws = (char*)d_ws;
    auto alloc = [&](size_t bytes) { char* p = ws; ws += (bytes + 255) & ~(size_t)255; return p; };
    unsigned short* h_bf    = (unsigned short*)alloc((size_t)M * D * 2);
    unsigned short* q_bf    = (unsigned short*)alloc((size_t)M * D * 2);
    unsigned short* k_bf    = (unsigned short*)alloc((size_t)M * D * 2);
    unsigned short* v_bf    = (unsigned short*)alloc((size_t)M * D * 2);
    unsigned short* attn_bf = (unsigned short*)alloc((size_t)M * D * 2);
    float*          x2      = (float*)alloc((size_t)M * D * 4);
    unsigned short* h2_bf   = (unsigned short*)alloc((size_t)M * D * 2);
    unsigned short* a_bf    = (unsigned short*)alloc((size_t)M * 4 * D * 2);
    unsigned short* Wqt     = (unsigned short*)alloc((size_t)H * HD * HD * 2);
    unsigned short* Wkt     = (unsigned short*)alloc((size_t)H * HD * HD * 2);
    unsigned short* Wvt     = (unsigned short*)alloc((size_t)H * HD * HD * 2);
    unsigned short* Wpt     = (unsigned short*)alloc((size_t)D * D * 2);
    unsigned short* W1t     = (unsigned short*)alloc((size_t)4 * D * D * 2);
    unsigned short* W2t     = (unsigned short*)alloc((size_t)D * 4 * D * 2);

    // weight convert + transpose ([N,K] f32 -> [K,N] bf16), once per call
    cvt_t_bf16_kernel<<<dim3(2, 2, 16), 256, 0, stream>>>(Wq, Wqt, HD, HD);
    cvt_t_bf16_kernel<<<dim3(2, 2, 16), 256, 0, stream>>>(Wk, Wkt, HD, HD);
    cvt_t_bf16_kernel<<<dim3(2, 2, 16), 256, 0, stream>>>(Wv, Wvt, HD, HD);
    cvt_t_bf16_kernel<<<dim3(32, 32, 1), 256, 0, stream>>>(Wp, Wpt, D, D);
    cvt_t_bf16_kernel<<<dim3(32, 128, 1), 256, 0, stream>>>(W1, W1t, 4 * D, D);
    cvt_t_bf16_kernel<<<dim3(128, 32, 1), 256, 0, stream>>>(W2, W2t, D, 4 * D);

    // LN1 -> h (bf16)
    ln_bf16_kernel<<<M, 256, 0, stream>>>(x, g1, be1, h_bf, D);

    // per-head QKV projections
    qkv_kernel<<<dim3(M / 128, H), 256, 0, stream>>>(h_bf, Wqt, Wkt, Wvt,
                                                     bq, bk, bv, q_bf, k_bf, v_bf);

    // causal flash attention
    attn_kernel<<<dim3(S / 64, B * H), 128, 0, stream>>>(q_bf, k_bf, v_bf, attn_bf);

    // out-proj + residual: x2 = x + attn @ Wp^T + bp
    gemm_kernel<0, false, true><<<dim3(M / 128, D / 256), 256, 0, stream>>>(
        attn_bf, Wpt, bp, x, x2, M, D, D);

    // LN2 -> h2 (bf16)
    ln_bf16_kernel<<<M, 256, 0, stream>>>(x2, g2, be2, h2_bf, D);

    // FFN1 + exact GeLU -> a (bf16)
    gemm_kernel<1, true, false><<<dim3(M / 128, 4 * D / 256), 256, 0, stream>>>(
        h2_bf, W1t, b1, nullptr, a_bf, M, 4 * D, D);

    // FFN2 + residual -> out
    gemm_kernel<0, false, true><<<dim3(M / 128, D / 256), 256, 0, stream>>>(
        a_bf, W2t, b2, x2, out, M, D, 4 * D);
}
